// MS_Attention_linear_59794534694971
// MI455X (gfx1250) — matrix operand order repaired and checked by reading, not compiled, not run on hardware
//
#include <hip/hip_runtime.h>
#include <hip/hip_bf16.h>

// MS_Attention_linear for MI455X (gfx1250, wave32, WMMA).
//
// Pipeline (all L2-resident, ~25MB traffic, ~4.6 GFLOP after exploiting
// (q k^T) v == q (k^T v), which is bit-exact here because spike() outputs
// are dyadic multiples of 1/4 and all partial sums stay on a 1/64 grid
// below 2^24):
//   1. bn_prep   : fold BN into per-channel (inv, beta) in f32
//   2. wcvt      : weights f32 -> f16 (spike inputs are exact in f16)
//   3. spike_x   : xs = spike(x), stored pixel-major [b][p][c] f16
//   4. gemm_conv : W x xs via v_wmma_f32_16x16x32_f16, BN(+spike) epilogue
//                  q -> pixel-major [b][p][c]; k,v -> head-T [b][h][d|e][p]
//   5. attn      : KV = k^T v (f16 WMMA, exact, ds_add_f32 cross-wave
//                  reduction in LDS), then out = q*KV*scale via f32 WMMA
//                  (KV values exceed f16's exact range), spike -> [b][p][c]
//   6. gemm_conv : proj conv + BN, f32 out in [b][o][p] (NCHW) to d_out

typedef _Float16 v16h __attribute__((ext_vector_type(16)));
typedef _Float16 v8h  __attribute__((ext_vector_type(8)));
typedef float    v8f  __attribute__((ext_vector_type(8)));
typedef float    v2f  __attribute__((ext_vector_type(2)));

#define B_ 8
#define C_ 256
#define N_ 1024
#define NH 8
#define HD 32

union AFrag { v16h v; v8h h[2]; };

__device__ __forceinline__ float spike4(float y) {
  // round(clamp(y,0,4))/4 ; jnp.round == round-half-even == rintf (RNE)
  float c = fminf(fmaxf(y, 0.f), 4.f);
  return rintf(c) * 0.25f;
}

// ---------------------------------------------------------------- prep ----
__global__ void bn_prep(const float* __restrict__ g, const float* __restrict__ b,
                        const float* __restrict__ m, const float* __restrict__ v,
                        float* __restrict__ inv, float* __restrict__ beta) {
  int c = threadIdx.x;
  float iv = g[c] / sqrtf(v[c] + 1e-5f);
  inv[c]  = iv;
  beta[c] = b[c] - m[c] * iv;
}

__global__ void wcvt(const float* __restrict__ w, _Float16* __restrict__ wh) {
  int i = blockIdx.x * blockDim.x + threadIdx.x;
  wh[i] = (_Float16)w[i];
}

__global__ void spike_x_kernel(const float* __restrict__ x, _Float16* __restrict__ xsT) {
  int i = blockIdx.x * blockDim.x + threadIdx.x;   // over B*C*N
  int p = i & (N_ - 1);
  int c = (i >> 10) & (C_ - 1);
  int b = i >> 18;
  xsT[(((size_t)b << 10) + p) * C_ + c] = (_Float16)spike4(x[i]);
}

// ------------------------------------------------------------- conv GEMM --
// y[o][p] = sum_c W[o][c] * act[p][c]    (per batch)
// One wave computes a 16(o) x 16(p) tile, K=256 in 8 steps of 32.
// A (weights, row-major): lanes 0-15 hold row o=lane, K={0..7,16..23};
// lanes 16-31 hold row o=lane-16, K={8..15,24..31}  -> two v8h loads.
// B (act, pixel-major so K=c is contiguous per pixel): lane holds column
// p=lane%16, same K sets as A                         -> two v8h loads.
// mode 0: spike -> f16 [b][p][c] ; mode 1: spike -> f16 [b][h][d][p]
// mode 2: f32 BN -> d_out [b][o][p]
__global__ __launch_bounds__(256) void gemm_conv(
    const _Float16* __restrict__ act, const _Float16* __restrict__ wgt,
    const float* __restrict__ inv, const float* __restrict__ beta,
    _Float16* __restrict__ outH, float* __restrict__ outF, int mode) {
  const int tid  = threadIdx.x;
  const int lane = tid & 31, wave = tid >> 5;
  const int t  = blockIdx.x * 8 + wave;       // 8192 tiles: [b][ot][pt]
  const int b  = t >> 10;
  const int ot = (t >> 6) & 15;
  const int pt = t & 63;
  const bool lo = (lane < 16);
  const int n = lane & 15;
  const int p = (pt << 4) + n;

  const _Float16* aRow = wgt + (((ot << 4) + n) * C_) + (lo ? 0 : 8);
  const _Float16* bRow = act + (((size_t)(b << 10) + p) * C_) + (lo ? 0 : 8);
  __builtin_prefetch(bRow, 0, 1);

  v8f acc = {};
#pragma unroll
  for (int kk = 0; kk < 8; ++kk) {
    const int k0 = kk * 32;
    AFrag a, bb;
    a.h[0]  = *(const v8h*)(aRow + k0);
    a.h[1]  = *(const v8h*)(aRow + k0 + 16);
    bb.h[0] = *(const v8h*)(bRow + k0);
    bb.h[1] = *(const v8h*)(bRow + k0 + 16);
    acc = __builtin_amdgcn_wmma_f32_16x16x32_f16(false, a.v, false, bb.v,
                                                 (short)0, acc, false, false);
  }

  // D layout: lanes 0-15 rows 0..7, lanes 16-31 rows 8..15 (col = lane%16)
  const int obase = (ot << 4) + (lo ? 0 : 8);
  if (mode == 0) {
    v8h out;
#pragma unroll
    for (int r = 0; r < 8; ++r) {
      int o = obase + r;
      out[r] = (_Float16)spike4(acc[r] * inv[o] + beta[o]);
    }
    *(v8h*)(outH + ((size_t)(b << 10) + p) * C_ + obase) = out;
  } else if (mode == 1) {
#pragma unroll
    for (int r = 0; r < 8; ++r) {
      int o = obase + r;
      int h = o >> 5, d = o & 31;
      outH[(((size_t)(b << 3) + h) * HD + d) * N_ + p] =
          (_Float16)spike4(acc[r] * inv[o] + beta[o]);
    }
  } else {
#pragma unroll
    for (int r = 0; r < 8; ++r) {
      int o = obase + r;
      outF[((size_t)(b << 8) + o) * N_ + p] = acc[r] * inv[o] + beta[o];
    }
  }
}

// ------------------------------------------------------------- attention --
// One workgroup (8 waves) per (b,h).
// Phase 1: KV[d][e] = sum_p k[p][d]*v[p][e]  (M=d,N=e,K=p=1024)
//   4 (mt,et) tiles x 2 K-halves across 8 waves; f16 WMMA; cross-wave
//   reduction via ds_add_f32 into LDS.
// Phase 2: out[p][e] = scale * sum_d q[p][d]*KV[d][e] via f32 WMMA
//   (KV magnitudes ~2^14 on a 1/16 grid are NOT exact in f16).
__global__ __launch_bounds__(256) void attn_kernel(
    const _Float16* __restrict__ qP,   // [b][p][C]
    const _Float16* __restrict__ kT,   // [b][h][d][p]
    const _Float16* __restrict__ vT,   // [b][h][e][p]
    _Float16* __restrict__ sP) {       // [b][p][C]
  __shared__ float KV[HD * HD];
  const int tid = threadIdx.x, lane = tid & 31, wave = tid >> 5;
  const int bh = blockIdx.x, b = bh >> 3, h = bh & 7;
  const bool lo = (lane < 16);
  const int n = lane & 15;

  for (int i = tid; i < HD * HD; i += 256) KV[i] = 0.f;
  __syncthreads();

  { // phase 1
    const int mt = wave & 1, et = (wave >> 1) & 1, kh = wave >> 2;
    const _Float16* kRow = kT + ((size_t)bh * HD + (mt << 4) + n) * N_ + (lo ? 0 : 8);
    const _Float16* vRow = vT + ((size_t)bh * HD + (et << 4) + n) * N_ + (lo ? 0 : 8);
    v8f acc = {};
#pragma unroll 4
    for (int s = 0; s < 16; ++s) {
      const int p0 = ((kh << 4) + s) << 5;
      AFrag a, bb;
      a.h[0]  = *(const v8h*)(kRow + p0);
      a.h[1]  = *(const v8h*)(kRow + p0 + 16);
      bb.h[0] = *(const v8h*)(vRow + p0);
      bb.h[1] = *(const v8h*)(vRow + p0 + 16);
      acc = __builtin_amdgcn_wmma_f32_16x16x32_f16(false, a.v, false, bb.v,
                                                   (short)0, acc, false, false);
    }
    const int m0 = (mt << 4) + (lo ? 0 : 8);
    const int e  = (et << 4) + n;
#pragma unroll
    for (int r = 0; r < 8; ++r) atomicAdd(&KV[(m0 + r) * HD + e], acc[r]);
  }
  __syncthreads();

  const float scale = 0.17677669529663687f;  // 1/sqrt(32)
  for (int pi = 0; pi < 8; ++pi) {
    const int pt = wave * 8 + pi;
#pragma unroll
    for (int et = 0; et < 2; ++et) {
      const int e = (et << 4) + n;
      float vals[8];
#if __has_builtin(__builtin_amdgcn_wmma_f32_16x16x4_f32)
      // f32 A 16x4: lanes 0-15 row=lane K={0,1}; lanes 16-31 row=lane-16 K={2,3}
      const _Float16* qp =
          qP + ((size_t)(b << 10) + (pt << 4) + n) * C_ + (h << 5) + (lo ? 0 : 2);
      v8f acc = {};
#pragma unroll
      for (int kd = 0; kd < 8; ++kd) {
        v2f a, bb;
        a.x = (float)qp[kd * 4];
        a.y = (float)qp[kd * 4 + 1];
        const int kr = kd * 4 + (lo ? 0 : 2);
        bb.x = KV[kr * HD + e];
        bb.y = KV[(kr + 1) * HD + e];
        acc = __builtin_amdgcn_wmma_f32_16x16x4_f32(false, a, false, bb,
                                                    (short)0, acc, false, false);
      }
#pragma unroll
      for (int r = 0; r < 8; ++r) vals[r] = acc[r];
#else
      const int m0f = lo ? 0 : 8;
#pragma unroll
      for (int r = 0; r < 8; ++r) {
        const _Float16* qr =
            qP + ((size_t)(b << 10) + (pt << 4) + m0f + r) * C_ + (h << 5);
        float y = 0.f;
        for (int d = 0; d < HD; ++d) y += (float)qr[d] * KV[d * HD + e];
        vals[r] = y;
      }
#endif
      const int m0 = lo ? 0 : 8;
#pragma unroll
      for (int r = 0; r < 8; ++r) {
        int pr = (pt << 4) + m0 + r;
        sP[((size_t)(b << 10) + pr) * C_ + (h << 5) + e] =
            (_Float16)spike4(vals[r] * scale);
      }
    }
  }
}

// ---------------------------------------------------------------- launch --
extern "C" void kernel_launch(void* const* d_in, const int* in_sizes, int n_in,
                              void* d_out, int out_size, void* d_ws, size_t ws_size,
                              hipStream_t stream) {
  const float* x = (const float*)d_in[0];
  const float *W[4], *G[4], *Bb[4], *M[4], *V[4];
  for (int s = 0; s < 4; ++s) {
    W[s]  = (const float*)d_in[1 + 5 * s + 0];
    G[s]  = (const float*)d_in[1 + 5 * s + 1];
    Bb[s] = (const float*)d_in[1 + 5 * s + 2];
    M[s]  = (const float*)d_in[1 + 5 * s + 3];
    V[s]  = (const float*)d_in[1 + 5 * s + 4];
  }

  size_t off = 0;
  auto carve = [&](size_t bytes) -> char* {
    char* p = (char*)d_ws + off;
    off += (bytes + 255) & ~(size_t)255;
    return p;
  };
  _Float16* wh  = (_Float16*)carve((size_t)4 * C_ * C_ * 2);  // f16 weights
  float*    bnI = (float*)carve(4 * C_ * 4);
  float*    bnB = (float*)carve(4 * C_ * 4);
  _Float16* xsT = (_Float16*)carve((size_t)B_ * N_ * C_ * 2); // [b][p][c]
  _Float16* qP  = (_Float16*)carve((size_t)B_ * N_ * C_ * 2); // [b][p][c]
  _Float16* kTT = (_Float16*)carve((size_t)B_ * N_ * C_ * 2); // [b][h][d][p]
  _Float16* vTT = (_Float16*)carve((size_t)B_ * N_ * C_ * 2); // [b][h][e][p]
  _Float16* sP  = xsT;  // xs is dead after the QKV GEMMs -> reuse

  for (int s = 0; s < 4; ++s) {
    bn_prep<<<1, 256, 0, stream>>>(G[s], Bb[s], M[s], V[s], bnI + s * C_, bnB + s * C_);
    wcvt<<<C_ * C_ / 256, 256, 0, stream>>>(W[s], wh + (size_t)s * C_ * C_);
  }
  spike_x_kernel<<<B_ * C_ * N_ / 256, 256, 0, stream>>>(x, xsT);

  gemm_conv<<<1024, 256, 0, stream>>>(xsT, wh + 0 * C_ * C_, bnI + 0,   bnB + 0,   qP,  nullptr, 0);
  gemm_conv<<<1024, 256, 0, stream>>>(xsT, wh + 1 * C_ * C_, bnI + 256, bnB + 256, kTT, nullptr, 1);
  gemm_conv<<<1024, 256, 0, stream>>>(xsT, wh + 2 * C_ * C_, bnI + 512, bnB + 512, vTT, nullptr, 1);

  attn_kernel<<<B_ * NH, 256, 0, stream>>>(qP, kTT, vTT, sP);

  gemm_conv<<<1024, 256, 0, stream>>>(sP, wh + 3 * C_ * C_, bnI + 768, bnB + 768,
                                      nullptr, (float*)d_out, 2);
}
